// MambaByteLM_20693152432813
// MI455X (gfx1250) — matrix operand order repaired and checked by reading, not compiled, not run on hardware
//
#include <hip/hip_runtime.h>

// ---------------- model constants (from reference) ----------------
#define T_LEN   1024
#define DMODEL  1024
#define E_DIM   2048
#define NLAYER  8
#define NSTATE  16
#define RRANK   64
#define VOCAB   256
#define XDBL_W  96          // R + 2N

typedef __attribute__((ext_vector_type(16))) __bf16 v16bf;
typedef __attribute__((ext_vector_type(2)))  __bf16 v2bf;
typedef __attribute__((ext_vector_type(8)))  float  v8f;
typedef __attribute__((ext_vector_type(4)))  int    v4i;

typedef __attribute__((address_space(1))) v4i* as1_v4i_ptr;
typedef __attribute__((address_space(3))) v4i* as3_v4i_ptr;

#define HAS_ASYNC_LDS (__has_builtin(__builtin_amdgcn_global_load_async_to_lds_b128) && \
                       __has_builtin(__builtin_amdgcn_s_wait_asynccnt))

__device__ __forceinline__ float silu_(float v) { return v / (1.f + __expf(-v)); }

// packs two f32 -> packed bf16 pair (v_cvt_pk_bf16_f32)
__device__ __forceinline__ unsigned pack_bf16(float lo, float hi) {
  v2bf p;
  p[0] = (__bf16)lo;
  p[1] = (__bf16)hi;
  return __builtin_bit_cast(unsigned, p);
}

// =====================================================================
// Embedding gather: h[t,d] = embed[ids[t], d]
// =====================================================================
__global__ __launch_bounds__(256)
void embed_kernel(const int* __restrict__ ids, const float* __restrict__ embed,
                  float* __restrict__ h)
{
  int i = blockIdx.x * 256 + threadIdx.x;     // over T*DM
  int t = i >> 10;                            // / DMODEL
  int d = i & (DMODEL - 1);
  h[i] = embed[(size_t)ids[t] * DMODEL + d];
}

// =====================================================================
// RMSNorm over last dim (DMODEL), one block (256 thr = 8 waves) per row.
// Row staged to LDS via global_load_async_to_lds_b128 so HBM sees it once.
// =====================================================================
__global__ __launch_bounds__(256)
void rmsnorm_kernel(const float* __restrict__ in, const float* __restrict__ w,
                    float* __restrict__ out)
{
  const int t   = blockIdx.x;
  const int tid = threadIdx.x;
  __shared__ float srow[DMODEL];
  __shared__ float red[256];

  const float* grow = in + (size_t)t * DMODEL;

#if HAS_ASYNC_LDS
  __builtin_amdgcn_global_load_async_to_lds_b128(
      (as1_v4i_ptr)(uintptr_t)(grow + tid * 4),
      (as3_v4i_ptr)(unsigned)(uintptr_t)&srow[tid * 4],
      0, 0);
  __builtin_amdgcn_s_wait_asynccnt(0);
#else
  *(float4*)&srow[tid * 4] = *(const float4*)&grow[tid * 4];
#endif
  __syncthreads();

  float s = 0.f;
  for (int d = tid; d < DMODEL; d += 256) {
    float v = srow[d];
    s += v * v;
  }
  red[tid] = s;
  __syncthreads();
  for (int off = 128; off > 0; off >>= 1) {
    if (tid < off) red[tid] += red[tid + off];
    __syncthreads();
  }
  float scale = rsqrtf(red[0] * (1.f / DMODEL) + 1e-5f);
  for (int d = tid; d < DMODEL; d += 256)
    out[(size_t)t * DMODEL + d] = srow[d] * scale * w[d];
}

// =====================================================================
// bf16 WMMA GEMM:  Cout[M,N] = A[M,K] @ B[K,N] (+ Cin)
// f32 operands converted to bf16 while staged into LDS in *fragment order*
// (ISA 7.12.2), so every operand fragment is 2x aligned ds_load_b128.
// 256 threads = 8 wave32s; BK=32; double-buffered LDS.
// Staging is two-phase (batch loads into registers, then convert+store) so
// the global loads pipeline behind partial loadcnt waits instead of
// serializing on a single recycled register pair.
// Template SPW = strips per wave (1 -> BM=128, 2 -> BM=256). The 4 B
// fragments are loaded once per K-step and reused across all strips.
// Requires: M % BM == 0, K % 32 == 0 (true for every shape here); N guarded.
// Fragment maps:
//   A (16x32): element (r,c) -> lane = 16*((c>>3)&1) + r, j = (c&7) + 8*(c>>4)
//   B (32x16): element (k,n) -> lane = 16*((k>>3)&1) + n, j = (k&7) + 8*(k>>4)
//   C/D f32  : col = lane&15, row = 8*(lane>>4) + j
// =====================================================================
template<int SPW>
__global__ __launch_bounds__(256)
void gemm_bf16_wmma(const float* __restrict__ A, int lda,
                    const float* __restrict__ B, int ldb,
                    const float* Cin,            // may be null (residual add)
                    float* Cout, int ldc,
                    int M, int N, int K)
{
  constexpr int BM = SPW * 128;
  const int bm   = blockIdx.y * BM;
  const int bn   = blockIdx.x * 64;
  const int tid  = threadIdx.x;
  const int wave = tid >> 5;        // 0..7 -> strips SPW*wave .. SPW*wave+SPW-1
  const int lane = tid & 31;

  // fragment-order, pair-packed: A = (8*SPW) strips * 256 words, B = 4 ntiles * 256 words
  __shared__ unsigned sA[2][SPW * 8 * 256];
  __shared__ unsigned sB[2][4 * 256];

  // per-thread invariants of the staging index decomposition
  const int a_cp   = tid & 15;                  // A k-pair index
  const int a_ln   = (((a_cp >> 2) & 1) << 4);  // + (r & 15) later
  const int a_word = (a_cp & 3) + ((a_cp >> 3) << 2);
  const int b_c    = tid & 63;                  // B column in tile
  const int b_gc   = bn + b_c;

  auto stageA = [&](unsigned* dst, int k0) {
    #pragma unroll
    for (int g = 0; g < SPW; ++g) {             // groups of 8 batched loads
      float2 vals[8];
      #pragma unroll
      for (int i = 0; i < 8; ++i) {
        int r = (((g * 8 + i) << 8 | tid) >> 4);
        vals[i] = *(const float2*)&A[(size_t)(bm + r) * lda + k0 + 2 * a_cp];
      }
      #pragma unroll
      for (int i = 0; i < 8; ++i) {
        int r = (((g * 8 + i) << 8 | tid) >> 4);
        dst[((r >> 4) << 8) + ((a_ln + (r & 15)) << 3) + a_word] =
            pack_bf16(vals[i].x, vals[i].y);
      }
    }
  };
  auto stageB = [&](unsigned* dst, int k0) {
    float lo[4], hi[4];
    #pragma unroll
    for (int i = 0; i < 4; ++i) {               // all 8 loads batched
      int kp = (i << 2) | (tid >> 6);
      lo[i] = 0.f; hi[i] = 0.f;
      if (b_gc < N) {
        lo[i] = B[(size_t)(k0 + 2 * kp) * ldb + b_gc];
        hi[i] = B[(size_t)(k0 + 2 * kp + 1) * ldb + b_gc];
      }
    }
    #pragma unroll
    for (int i = 0; i < 4; ++i) {
      int kp = (i << 2) | (tid >> 6);
      int ln = (((kp >> 2) & 1) << 4) + (b_c & 15);
      dst[((b_c >> 4) << 8) + (ln << 3) + (kp & 3) + ((kp >> 3) << 2)] = pack_bf16(lo[i], hi[i]);
    }
  };

  v8f acc[SPW * 4];
  const v8f vzero = {0.f, 0.f, 0.f, 0.f, 0.f, 0.f, 0.f, 0.f};
  #pragma unroll
  for (int i = 0; i < SPW * 4; i++) acc[i] = vzero;

  stageA(sA[0], 0);
  stageB(sB[0], 0);

  int buf = 0;
  for (int k0 = 0; k0 < K; k0 += 32) {
    __syncthreads();
    if (k0 + 32 < K) {              // stage next K-tile into the other buffer
      stageA(sA[buf ^ 1], k0 + 32);
      stageB(sB[buf ^ 1], k0 + 32);
    }
    // preload ALL operand fragments, then back-to-back WMMAs
    const unsigned* pa = sA[buf] + wave * (SPW * 256);
    const unsigned* pb = sB[buf];
    v16bf a[SPW];
    #pragma unroll
    for (int s = 0; s < SPW; s++) a[s] = ((const v16bf*)(pa + s * 256))[lane];
    v16bf b[4];
    #pragma unroll
    for (int nt = 0; nt < 4; nt++) b[nt] = ((const v16bf*)(pb + nt * 256))[lane];
    #pragma unroll
    for (int s = 0; s < SPW; s++) {
      #pragma unroll
      for (int nt = 0; nt < 4; nt++) {
        acc[s * 4 + nt] = __builtin_amdgcn_wmma_f32_16x16x32_bf16(
            false, a[s], false, b[nt], (short)0, acc[s * 4 + nt], false, false);
      }
    }
    buf ^= 1;
  }

  // ---- store: row = bm + 16*(SPW*wave+s) + 8*(lane>=16) + j ; col = bn + 16*nt + (lane&15)
  const int r16   = lane & 15;
  const int mbase = (lane >> 4) * 8;
  #pragma unroll
  for (int s = 0; s < SPW; s++) {
    #pragma unroll
    for (int nt = 0; nt < 4; nt++) {
      int col = bn + nt * 16 + r16;
      if (col >= N) continue;
      #pragma unroll
      for (int j = 0; j < 8; j++) {
        int row = bm + (SPW * wave + s) * 16 + mbase + j;
        size_t idx = (size_t)row * ldc + col;
        float v = acc[s * 4 + nt][j];
        if (Cin) v += Cin[idx];
        Cout[idx] = v;
      }
    }
  }
}

// =====================================================================
// Causal depthwise conv (K=4) + bias + SiLU.
// x-part lives in first E columns of xz (row width 2E). Output x[T,E].
// =====================================================================
__global__ __launch_bounds__(256)
void conv_silu_kernel(const float* __restrict__ xz, const float* __restrict__ cw,
                      const float* __restrict__ cb, float* __restrict__ x)
{
  int i = blockIdx.x * 256 + threadIdx.x;     // over T*E
  int t = i >> 11;                            // / E_DIM
  int e = i & (E_DIM - 1);
  float acc = cb[e];
  #pragma unroll
  for (int k = 0; k < 4; k++) {
    int ts = t - 3 + k;
    if (ts >= 0) acc += cw[e * 4 + k] * xz[(size_t)ts * (2 * E_DIM) + e];
  }
  x[i] = silu_(acc);
}

// =====================================================================
// dt = softplus(dtraw + dt_bias[e])  (in place on dtraw)
// =====================================================================
__global__ __launch_bounds__(256)
void softplus_bias_kernel(float* __restrict__ dt, const float* __restrict__ bias)
{
  int i = blockIdx.x * 256 + threadIdx.x;     // over T*E
  int e = i & (E_DIM - 1);
  float v = dt[i] + bias[e];
  dt[i] = (v > 20.f) ? v : log1pf(__expf(v));
}

// =====================================================================
// Selective scan. Thread per (e, n): 32768 threads, sequential over T.
// n = lane bits [3:0] -> y reduction over n via 4x shfl_xor in half-wave.
// Fuses +D*x and the silu(z) gate; z = xz[:, E:2E].
// =====================================================================
__global__ __launch_bounds__(256)
void scan_kernel(const float* __restrict__ x,     // T x E (post conv/silu)
                 const float* __restrict__ dt,    // T x E (softplus applied)
                 const float* __restrict__ xdbl,  // T x 96 (B at 64, C at 80)
                 const float* __restrict__ A_log, // E x N (layer slice)
                 const float* __restrict__ Dp,    // E
                 const float* __restrict__ xz,    // T x 2E (z half)
                 float* __restrict__ y)           // T x E
{
  int idx = blockIdx.x * 256 + threadIdx.x;       // over E*NSTATE
  int e = idx >> 4;
  int n = idx & 15;
  float Aen = -__expf(A_log[e * NSTATE + n]);
  float De  = Dp[e];
  float h   = 0.f;
  for (int t = 0; t < T_LEN; t++) {
    float dte = dt[(size_t)t * E_DIM + e];
    float xe  = x [(size_t)t * E_DIM + e];
    float Bt  = xdbl[(size_t)t * XDBL_W + RRANK + n];
    float Ct  = xdbl[(size_t)t * XDBL_W + RRANK + NSTATE + n];
    h = __expf(dte * Aen) * h + dte * Bt * xe;
    float v = h * Ct;
    v += __shfl_xor(v, 1, 32);
    v += __shfl_xor(v, 2, 32);
    v += __shfl_xor(v, 4, 32);
    v += __shfl_xor(v, 8, 32);
    if (n == 0) {
      float ze = xz[(size_t)t * (2 * E_DIM) + E_DIM + e];
      y[(size_t)t * E_DIM + e] = (v + De * xe) * silu_(ze);
    }
  }
}

// =====================================================================
// Host orchestration
// =====================================================================
static inline void launch_gemm(const float* A, int lda, const float* B, int ldb,
                               const float* Cin, float* Cout, int ldc,
                               int M, int N, int K, hipStream_t s)
{
  if (N >= 2048) {
    // wide output: maximize B-fragment reuse (BM=256)
    dim3 grid((N + 63) / 64, M / 256);
    gemm_bf16_wmma<2><<<grid, dim3(256), 0, s>>>(A, lda, B, ldb, Cin, Cout, ldc, M, N, K);
  } else {
    // narrow output: BM=128 doubles workgroup count for better fill
    dim3 grid((N + 63) / 64, M / 128);
    gemm_bf16_wmma<1><<<grid, dim3(256), 0, s>>>(A, lda, B, ldb, Cin, Cout, ldc, M, N, K);
  }
}

extern "C" void kernel_launch(void* const* d_in, const int* in_sizes, int n_in,
                              void* d_out, int out_size, void* d_ws, size_t ws_size,
                              hipStream_t stream)
{
  const int*   ids     = (const int*)  d_in[0];
  const float* embed   = (const float*)d_in[1];
  const float* Wxz     = (const float*)d_in[2];   // NL x DM x 2E
  const float* conv_w  = (const float*)d_in[3];   // NL x E x 4
  const float* conv_b  = (const float*)d_in[4];   // NL x E
  const float* Wx      = (const float*)d_in[5];   // NL x E x 96
  const float* Wdt     = (const float*)d_in[6];   // NL x R x E
  const float* dt_bias = (const float*)d_in[7];   // NL x E
  const float* A_log   = (const float*)d_in[8];   // NL x E x N
  const float* Dp      = (const float*)d_in[9];   // NL x E
  const float* Wout    = (const float*)d_in[10];  // NL x E x DM
  const float* norm_w  = (const float*)d_in[11];  // NL x DM
  const float* fnorm_w = (const float*)d_in[12];  // DM
  const float* lm_head = (const float*)d_in[13];  // DM x V
  float* out = (float*)d_out;

  // ---- workspace layout (f32) ----
  float* h    = (float*)d_ws;                    // T*DM
  float* u    = h    + (size_t)T_LEN * DMODEL;   // T*DM
  float* xz   = u    + (size_t)T_LEN * DMODEL;   // T*4096
  float* x    = xz   + (size_t)T_LEN * 2 * E_DIM;// T*E
  float* xdbl = x    + (size_t)T_LEN * E_DIM;    // T*96
  float* dt   = xdbl + (size_t)T_LEN * XDBL_W;   // T*E
  float* y    = dt   + (size_t)T_LEN * E_DIM;    // T*E

  embed_kernel<<<(T_LEN * DMODEL) / 256, 256, 0, stream>>>(ids, embed, h);

  for (int l = 0; l < NLAYER; l++) {
    const float* lWxz  = Wxz     + (size_t)l * DMODEL * 2 * E_DIM;
    const float* lcw   = conv_w  + (size_t)l * E_DIM * 4;
    const float* lcb   = conv_b  + (size_t)l * E_DIM;
    const float* lWx   = Wx      + (size_t)l * E_DIM * XDBL_W;
    const float* lWdt  = Wdt     + (size_t)l * RRANK * E_DIM;
    const float* ldtb  = dt_bias + (size_t)l * E_DIM;
    const float* lAlog = A_log   + (size_t)l * E_DIM * NSTATE;
    const float* lD    = Dp      + (size_t)l * E_DIM;
    const float* lWout = Wout    + (size_t)l * E_DIM * DMODEL;
    const float* lnw   = norm_w  + (size_t)l * DMODEL;

    // u = rmsnorm(h) * nw
    rmsnorm_kernel<<<T_LEN, 256, 0, stream>>>(h, lnw, u);
    // xz = u @ Wxz                      (1024 x 1024 x 4096)
    launch_gemm(u, DMODEL, lWxz, 2 * E_DIM, nullptr, xz, 2 * E_DIM,
                T_LEN, 2 * E_DIM, DMODEL, stream);
    // x = silu(causal_conv(xz[:, :E]) + cb)
    conv_silu_kernel<<<(T_LEN * E_DIM) / 256, 256, 0, stream>>>(xz, lcw, lcb, x);
    // xdbl = x @ Wx                     (1024 x 2048 x 96)
    launch_gemm(x, E_DIM, lWx, XDBL_W, nullptr, xdbl, XDBL_W,
                T_LEN, XDBL_W, E_DIM, stream);
    // dt_raw = xdbl[:, :R] @ Wdt        (1024 x 64 x 2048)
    launch_gemm(xdbl, XDBL_W, lWdt, E_DIM, nullptr, dt, E_DIM,
                T_LEN, E_DIM, RRANK, stream);
    // dt = softplus(dt_raw + dt_bias)
    softplus_bias_kernel<<<(T_LEN * E_DIM) / 256, 256, 0, stream>>>(dt, ldtb);
    // y = selective_scan(...) * silu(z)
    scan_kernel<<<(E_DIM * NSTATE) / 256, 256, 0, stream>>>(
        x, dt, xdbl, lAlog, lD, xz, y);
    // h = h + y @ Wout                  (1024 x 2048 x 1024, fused residual)
    launch_gemm(y, E_DIM, lWout, DMODEL, h, h, DMODEL,
                T_LEN, DMODEL, E_DIM, stream);
  }

  // final rmsnorm + lm_head
  rmsnorm_kernel<<<T_LEN, 256, 0, stream>>>(h, fnorm_w, u);
  launch_gemm(u, DMODEL, lm_head, VOCAB, nullptr, out, VOCAB,
              T_LEN, VOCAB, DMODEL, stream);
}
